// Block_49529562858265
// MI455X (gfx1250) — compile-verified, matrix operand order checked by reading, not run on hardware
//
#include <hip/hip_runtime.h>
#include <stdint.h>

// ---------------------------------------------------------------------------
// Transformer encoder block for MI455X (gfx1250, wave32, WMMA bf16)
// B=8 N=1024 C=768 H=12 HD=64 HID=3072
// GEMM operand staging: Tensor Data Mover (tensor_load_to_lds, TENSORcnt),
// attention K/V staging: GLOBAL_LOAD_ASYNC_TO_LDS_B128 (ASYNCcnt).
// ---------------------------------------------------------------------------

typedef __bf16 bf16;
typedef __attribute__((ext_vector_type(16))) __bf16 v16bf;
typedef __attribute__((ext_vector_type(8)))  __bf16 v8bf;
typedef __attribute__((ext_vector_type(8)))  float  v8f;
typedef __attribute__((ext_vector_type(4)))  unsigned int u32x4;
typedef __attribute__((ext_vector_type(8)))  unsigned int u32x8;

#define B_    8
#define N_    1024
#define C_    768
#define H_    12
#define HD_   64
#define HID_  3072
#define ROWS_ (B_ * N_)

static __device__ __forceinline__ v16bf mk16(v8bf lo, v8bf hi) {
    v16bf r;
#pragma unroll
    for (int e = 0; e < 8; ++e) { r[e] = lo[e]; r[e + 8] = hi[e]; }
    return r;
}

static __device__ __forceinline__ v8f wmma_bf16(v16bf a, v16bf b, v8f c) {
    return __builtin_amdgcn_wmma_f32_16x16x32_bf16(
        /*neg_a=*/false, a, /*neg_b=*/false, b,
        /*c_mod=*/(short)0, c, /*reuse_a=*/false, /*reuse_b=*/false);
}

// Flat LDS pointer -> 32-bit LDS byte offset (ISA 10.2: LDS aperture uses
// addr[31:0] truncated, so the low 32 flat bits are the LDS address).
static __device__ __forceinline__ unsigned ldsoff(const void* p) {
    return (unsigned)(uintptr_t)p;
}

// Async 16B copy global -> LDS, tracked by ASYNCcnt (no VGPR transit).
static __device__ __forceinline__ void async_b128(unsigned lds, const void* g) {
    asm volatile("global_load_async_to_lds_b128 %0, %1, off"
                 :: "v"(lds), "v"((unsigned long long)(uintptr_t)g)
                 : "memory");
}

#if __has_builtin(__builtin_amdgcn_s_wait_asynccnt)
#define WAIT_ASYNC(n) __builtin_amdgcn_s_wait_asynccnt(n)
#else
#define WAIT_ASYNC(n) asm volatile("s_wait_asynccnt " #n ::: "memory")
#endif

#if __has_builtin(__builtin_amdgcn_s_wait_tensorcnt)
#define WAIT_TENSOR(n) __builtin_amdgcn_s_wait_tensorcnt(n)
#else
#define WAIT_TENSOR(n) asm volatile("s_wait_tensorcnt " #n ::: "memory")
#endif

// ---------------------------------------------------------------------------
// TDM: one descriptor moves a 2D tile (tile_w x tile_h, 2-byte elems) from
// global memory into LDS, with per-row LDS padding (pad_interval=3 -> pad
// every 16 DWORDs = one 64B row; pad_amount=3 -> insert 4 DWORDs = 16B),
// i.e. exactly the 40-half padded row stride the WMMA frag readers use.
// D# bit layout per CDNA5 ISA 8.3/8.4. Tracked by TENSORcnt.
// ---------------------------------------------------------------------------
static __device__ __forceinline__ void tdm_load_2d_pad(
    unsigned lds_byte, const void* gaddr,
    unsigned tile_w_elems, unsigned tile_h_rows,
    unsigned tensor_w_elems, unsigned tensor_h_rows,
    unsigned row_stride_elems)
{
    unsigned long long ga = (unsigned long long)(uintptr_t)gaddr;
    u32x4 g0;
    g0[0] = 1u;                                           // count=1, user D#
    g0[1] = lds_byte;                                     // lds_addr
    g0[2] = (unsigned)(ga & 0xffffffffu);                 // global_addr[31:0]
    g0[3] = (unsigned)((ga >> 32) & 0x01ffffffu)          // global_addr[56:32]
          | (2u << 30);                                   // type = 2 (image)
    u32x8 g1;
    g1[0] = (1u << 16)        // data_size = 1 -> 2-byte elements
          | (1u << 20)        // pad_enable
          | (3u << 22)        // pad_interval: 16 DWORDs
          | (3u << 25);       // pad_amount:   4 DWORDs
    g1[1] = (tensor_w_elems & 0xffffu) << 16;             // tensor_dim0 lo
    g1[2] = ((tensor_w_elems >> 16) & 0xffffu)            // tensor_dim0 hi
          | ((tensor_h_rows & 0xffffu) << 16);            // tensor_dim1 lo
    g1[3] = ((tensor_h_rows >> 16) & 0xffffu)             // tensor_dim1 hi
          | ((tile_w_elems & 0xffffu) << 16);             // tile_dim0
    g1[4] = (tile_h_rows & 0xffffu);                      // tile_dim1 (dim2=0)
    g1[5] = row_stride_elems;                             // dim0_stride lo
    g1[6] = 0u;                                           // dim0_stride hi
    g1[7] = 0u;
    asm volatile("tensor_load_to_lds %0, %1" :: "s"(g0), "s"(g1) : "memory");
}

// ---------------------------------------------------------------------------
// Weight convert + transpose: W[K, Ncols] f32  ->  WT[Ncols, K] bf16
// ---------------------------------------------------------------------------
__global__ __launch_bounds__(256) void cvt_transpose_kernel(
    const float* __restrict__ W, bf16* __restrict__ WT, int K, int Ncols)
{
    int idx = blockIdx.x * 256 + threadIdx.x;
    int n  = idx / K;
    int kk = idx - n * K;
    WT[idx] = (bf16)W[(size_t)kk * Ncols + n];
}

// ---------------------------------------------------------------------------
// LayerNorm over C=768, one row per block (256 thr = 8 waves), bf16 output
// ---------------------------------------------------------------------------
__global__ __launch_bounds__(256) void ln_bf16_kernel(
    const float* __restrict__ x, const float* __restrict__ g,
    const float* __restrict__ b, bf16* __restrict__ y)
{
    const int row = blockIdx.x;
    const int t   = threadIdx.x;
    const float* xr = x + (size_t)row * C_;
    float v0 = xr[t], v1 = xr[t + 256], v2 = xr[t + 512];
    float s = v0 + v1 + v2;
    float q = v0 * v0 + v1 * v1 + v2 * v2;
#pragma unroll
    for (int m = 16; m >= 1; m >>= 1) {
        s += __shfl_xor(s, m);
        q += __shfl_xor(q, m);
    }
    __shared__ float ss[8], qq[8];
    if ((t & 31) == 0) { ss[t >> 5] = s; qq[t >> 5] = q; }
    __syncthreads();
    float S = 0.f, Q = 0.f;
#pragma unroll
    for (int i = 0; i < 8; ++i) { S += ss[i]; Q += qq[i]; }
    float mu   = S * (1.0f / C_);
    float var  = Q * (1.0f / C_) - mu * mu;
    float rstd = rsqrtf(var + 1e-5f);
    bf16* yr = y + (size_t)row * C_;
    yr[t]       = (bf16)((v0 - mu) * rstd * g[t]       + b[t]);
    yr[t + 256] = (bf16)((v1 - mu) * rstd * g[t + 256] + b[t + 256]);
    yr[t + 512] = (bf16)((v2 - mu) * rstd * g[t + 512] + b[t + 512]);
}

// ---------------------------------------------------------------------------
// bf16 GEMM: C[M,Ncols] = A[M,K] * BT[Ncols,K]^T, 128x128x32 block tiles,
// 8 waves (2x4), wave tile 64x32, v_wmma_f32_16x16x32_bf16 core.
// Double-buffered TDM (tensor_load_to_lds) tile staging, wave 0 issues.
//   MODE 0: QKV epilogue  -> q[B,H,N,HD], k[B,H,N,HD], vT[B,H,HD,N] (bf16)
//   MODE 1: resid+bias f32 -> outf = resid + acc + bias
//   MODE 2: gelu(acc+bias) -> outb (bf16)
// ---------------------------------------------------------------------------
template<int MODE>
__global__ __launch_bounds__(256) void gemm_bf16_kernel(
    const bf16* __restrict__ A, const bf16* __restrict__ BT, int K, int Ncols,
    const float* __restrict__ bias, const float* __restrict__ resid,
    float* __restrict__ outf, bf16* __restrict__ outb,
    bf16* __restrict__ qp, bf16* __restrict__ kp, bf16* __restrict__ vp)
{
    __shared__ bf16 aT[2][128 * 40];   // 128 rows x (32 + 8 pad) halfs
    __shared__ bf16 bT[2][128 * 40];

    const int tid    = threadIdx.x;
    const int lane   = tid & 31;
    const int wave   = tid >> 5;
    const int lh     = lane >> 4;   // lane-half per WMMA 16-bit A layout
    const int lm     = lane & 15;
    const int wm     = wave >> 2;   // 0..1 over M
    const int wn     = wave & 3;    // 0..3 over N
    const int blockM = blockIdx.y * 128;
    const int blockN = blockIdx.x * 128;

    v8f acc[4][2];
#pragma unroll
    for (int i = 0; i < 4; ++i)
#pragma unroll
        for (int j = 0; j < 2; ++j) acc[i][j] = (v8f){};

    const bf16* aTile = A  + (size_t)blockM * K;   // tile row 0, col kt
    const bf16* bTile = BT + (size_t)blockN * K;
    const unsigned Ku = (unsigned)K;

    auto issue = [&](int kt, int buf) {
        tdm_load_2d_pad(ldsoff(&aT[buf][0]), aTile + kt,
                        32u, 128u, Ku, 1u << 20, Ku);
        tdm_load_2d_pad(ldsoff(&bT[buf][0]), bTile + kt,
                        32u, 128u, Ku, 1u << 20, Ku);
    };

    const int nk = K >> 5;
    if (wave == 0) issue(0, 0);
    for (int t = 0; t < nk; ++t) {
        if (wave == 0) {
            if (t + 1 < nk) {
                issue((t + 1) << 5, (t + 1) & 1);   // next tile via TDM
                WAIT_TENSOR(2);                     // current buffer complete
            } else {
                WAIT_TENSOR(0);
            }
        }
        __syncthreads();

        const bf16* aL = aT[t & 1];
        const bf16* bL = bT[t & 1];
        v16bf af[4], bfr[2];
#pragma unroll
        for (int i = 0; i < 4; ++i) {
            int row = wm * 64 + i * 16 + lm;
            v8bf lo = *(const v8bf*)&aL[row * 40 + 8 * lh];
            v8bf hi = *(const v8bf*)&aL[row * 40 + 16 + 8 * lh];
            af[i] = mk16(lo, hi);
        }
#pragma unroll
        for (int j = 0; j < 2; ++j) {
            int row = wn * 32 + j * 16 + lm;
            v8bf lo = *(const v8bf*)&bL[row * 40 + 8 * lh];
            v8bf hi = *(const v8bf*)&bL[row * 40 + 16 + 8 * lh];
            bfr[j] = mk16(lo, hi);
        }
#pragma unroll
        for (int i = 0; i < 4; ++i)
#pragma unroll
            for (int j = 0; j < 2; ++j)
                acc[i][j] = wmma_bf16(af[i], bfr[j], acc[i][j]);
        __syncthreads();   // all waves done reading before next TDM write
    }

    // ---- epilogue: D layout (lane<16: M=g, N=lane; lane>=16: M=g+8) ----
#pragma unroll
    for (int i = 0; i < 4; ++i) {
#pragma unroll
        for (int j = 0; j < 2; ++j) {
            int d = blockN + wn * 32 + j * 16 + lm;
#pragma unroll
            for (int gEl = 0; gEl < 8; ++gEl) {
                int r = blockM + wm * 64 + i * 16 + gEl + 8 * lh;
                float v = acc[i][j][gEl];
                if (MODE == 0) {
                    int bb = r >> 10, n = r & 1023;
                    int which = d / C_;
                    int c = d - which * C_;
                    int h = c >> 6, hd = c & 63;
                    bf16 val = (bf16)v;
                    if (which == 0)
                        qp[(((size_t)(bb * H_ + h)) * N_ + n) * HD_ + hd] = val;
                    else if (which == 1)
                        kp[(((size_t)(bb * H_ + h)) * N_ + n) * HD_ + hd] = val;
                    else
                        vp[((size_t)(bb * H_ + h) * HD_ + hd) * N_ + n] = val;
                } else if (MODE == 1) {
                    size_t o = (size_t)r * Ncols + d;
                    outf[o] = resid[o] + v + bias[d];
                } else {
                    float tv = v + bias[d];
                    outb[(size_t)r * Ncols + d] =
                        (bf16)(0.5f * tv * (1.0f + erff(tv * 0.70710678f)));
                }
            }
        }
    }
}

// ---------------------------------------------------------------------------
// Flash attention: 4 waves/WG = 4 consecutive 16-row Q tiles of the SAME
// (b,h) -> K/V tiles staged once per WG in LDS (async, double-buffered) and
// shared by all 4 waves. 32 keys per step, HD=64. Online softmax; P goes
// D-layout -> A-layout through a 1KB per-wave LDS tile.
// ---------------------------------------------------------------------------
__global__ __launch_bounds__(128) void attn_kernel(
    const bf16* __restrict__ q, const bf16* __restrict__ k,
    const bf16* __restrict__ vT, bf16* __restrict__ outp)
{
    __shared__ bf16 klds[2][32 * 72];   // 32 keys x (64 + 8 pad) halfs
    __shared__ bf16 vlds[2][64 * 40];   // 64 hd   x (32 + 8 pad) halfs
    __shared__ bf16 plds[4][16 * 40];

    const int tid  = threadIdx.x;
    const int lane = tid & 31;
    const int wid  = tid >> 5;
    const int lh   = lane >> 4;
    const int lm   = lane & 15;

    const int tile = blockIdx.x * 4 + wid;       // B*H*(N/16) tiles
    const int q0   = (tile & 63) * 16;           // N/16 = 64
    const int bh   = tile >> 6;                  // same for all 4 waves
    const int bb   = bh / H_;
    const int h    = bh - bb * H_;

    const bf16* qb = q  + (size_t)bh * N_ * HD_;
    const bf16* kb = k  + (size_t)bh * N_ * HD_;
    const bf16* vb = vT + (size_t)bh * HD_ * N_;
    bf16* pl = plds[wid];

    // per-thread loader mapping for the shared K/V tiles
    const int krow = tid >> 2, kcg = (tid & 3) * 16;   // K: 32 x 64
    const int vrow = tid >> 1, vcg = (tid & 1) * 16;   // V: 64 x 32

    auto issue = [&](int j, int buf) {
        const bf16* kg = kb + (size_t)(j * 32 + krow) * HD_ + kcg;
        const bf16* vg = vb + (size_t)vrow * N_ + j * 32 + vcg;
        unsigned lk = ldsoff(&klds[buf][krow * 72 + kcg]);
        unsigned lv = ldsoff(&vlds[buf][vrow * 40 + vcg]);
        async_b128(lk,      kg);
        async_b128(lk + 16, kg + 8);
        async_b128(lv,      vg);
        async_b128(lv + 16, vg + 8);
    };

    // Q fragments (16 x 64 -> two K=32 frags), loaded once from global
    v16bf qf[2];
    {
        const bf16* qr = qb + (size_t)(q0 + lm) * HD_;
#pragma unroll
        for (int f = 0; f < 2; ++f) {
            v8bf lo = *(const v8bf*)(qr + f * 32 + 8 * lh);
            v8bf hi = *(const v8bf*)(qr + f * 32 + 16 + 8 * lh);
            qf[f] = mk16(lo, hi);
        }
    }

    const float kscale = 0.125f * 1.44269504f;   // HD^-0.5 * log2(e)

    float mr[8], lr_[8];
    v8f o4[4];
#pragma unroll
    for (int gEl = 0; gEl < 8; ++gEl) { mr[gEl] = -1e30f; lr_[gEl] = 0.f; }
#pragma unroll
    for (int t = 0; t < 4; ++t) o4[t] = (v8f){};

    const int nj = N_ / 32;
    issue(0, 0);
    for (int j = 0; j < nj; ++j) {
        if (j + 1 < nj) {
            issue(j + 1, (j + 1) & 1);
            WAIT_ASYNC(4);
        } else {
            WAIT_ASYNC(0);
        }
        __syncthreads();

        const bf16* kL = klds[j & 1];
        const bf16* vL = vlds[j & 1];

        // K fragments from LDS (B^T = K rows, row-major, stride 72)
        v16bf kf[2][2];
#pragma unroll
        for (int nc = 0; nc < 2; ++nc) {
            const bf16* kr = kL + (nc * 16 + lm) * 72;
#pragma unroll
            for (int f = 0; f < 2; ++f) {
                v8bf lo = *(const v8bf*)(kr + f * 32 + 8 * lh);
                v8bf hi = *(const v8bf*)(kr + f * 32 + 16 + 8 * lh);
                kf[nc][f] = mk16(lo, hi);
            }
        }
        v8f s0 = (v8f){}, s1 = (v8f){};
        s0 = wmma_bf16(qf[0], kf[0][0], s0);
        s0 = wmma_bf16(qf[1], kf[0][1], s0);
        s1 = wmma_bf16(qf[0], kf[1][0], s1);
        s1 = wmma_bf16(qf[1], kf[1][1], s1);

        // online softmax per row (row = g + 8*lh, replicated over 16 lanes)
#pragma unroll
        for (int gEl = 0; gEl < 8; ++gEl) {
            float tmax = fmaxf(s0[gEl], s1[gEl]);
#pragma unroll
            for (int m = 8; m >= 1; m >>= 1) tmax = fmaxf(tmax, __shfl_xor(tmax, m));
            float nm    = fmaxf(mr[gEl], tmax);
            float alpha = exp2f((mr[gEl] - nm) * kscale);
            mr[gEl] = nm;
            float p0 = exp2f((s0[gEl] - nm) * kscale);
            float p1 = exp2f((s1[gEl] - nm) * kscale);
            float rs = p0 + p1;
#pragma unroll
            for (int m = 8; m >= 1; m >>= 1) rs += __shfl_xor(rs, m);
            lr_[gEl] = lr_[gEl] * alpha + rs;
#pragma unroll
            for (int t = 0; t < 4; ++t) o4[t][gEl] *= alpha;
            int mrow = gEl + 8 * lh;
            pl[mrow * 40 + lm]      = (bf16)p0;
            pl[mrow * 40 + 16 + lm] = (bf16)p1;
        }

        // P: D-layout -> A-layout through per-wave LDS tile
        v16bf pf;
        {
            v8bf lo = *(const v8bf*)&pl[lm * 40 + 8 * lh];
            v8bf hi = *(const v8bf*)&pl[lm * 40 + 16 + 8 * lh];
            pf = mk16(lo, hi);
        }

        // O += P * V  (B^T = V^T rows, stride 40 in LDS)
#pragma unroll
        for (int t = 0; t < 4; ++t) {
            const bf16* vr = vL + (t * 16 + lm) * 40;
            v8bf lo = *(const v8bf*)(vr + 8 * lh);
            v8bf hi = *(const v8bf*)(vr + 16 + 8 * lh);
            o4[t] = wmma_bf16(pf, mk16(lo, hi), o4[t]);
        }
        __syncthreads();   // all waves done with this buffer
    }

    // normalize and emit bf16 into [B, N, C] at column h*64
#pragma unroll
    for (int gEl = 0; gEl < 8; ++gEl) {
        float inv = 1.0f / lr_[gEl];
        int qg = q0 + gEl + 8 * lh;
        bf16* orow = outp + ((size_t)bb * N_ + qg) * C_ + h * HD_;
#pragma unroll
        for (int t = 0; t < 4; ++t)
            orow[t * 16 + lm] = (bf16)(o4[t][gEl] * inv);
    }
}

// ---------------------------------------------------------------------------
// Host orchestration
// ---------------------------------------------------------------------------
extern "C" void kernel_launch(void* const* d_in, const int* in_sizes, int n_in,
                              void* d_out, int out_size, void* d_ws, size_t ws_size,
                              hipStream_t stream)
{
    const float* x      = (const float*)d_in[0];
    const float* ln1_g  = (const float*)d_in[1];
    const float* ln1_b  = (const float*)d_in[2];
    const float* w_qkv  = (const float*)d_in[3];
    const float* w_proj = (const float*)d_in[4];
    const float* b_proj = (const float*)d_in[5];
    const float* ln2_g  = (const float*)d_in[6];
    const float* ln2_b  = (const float*)d_in[7];
    const float* w_fc1  = (const float*)d_in[8];
    const float* b_fc1  = (const float*)d_in[9];
    const float* w_fc2  = (const float*)d_in[10];
    const float* b_fc2  = (const float*)d_in[11];
    float* out = (float*)d_out;

    char*  ws  = (char*)d_ws;
    size_t off = 0;
    auto take = [&](size_t bytes) -> char* {
        char* p = ws + off;
        off = (off + bytes + 255) & ~(size_t)255;
        return p;
    };
    bf16*  wqkvT  = (bf16*)take((size_t)3 * C_ * C_ * 2);    // [3C, C]
    bf16*  wprojT = (bf16*)take((size_t)C_ * C_ * 2);        // [C, C]
    bf16*  wfc1T  = (bf16*)take((size_t)HID_ * C_ * 2);      // [HID, C]
    bf16*  wfc2T  = (bf16*)take((size_t)C_ * HID_ * 2);      // [C, HID]
    bf16*  h1     = (bf16*)take((size_t)ROWS_ * C_ * 2);     // ln1 out
    bf16*  qbuf   = (bf16*)take((size_t)ROWS_ * C_ * 2);     // [B,H,N,HD]
    bf16*  kbuf   = (bf16*)take((size_t)ROWS_ * C_ * 2);     // [B,H,N,HD]
    bf16*  vbufT  = (bf16*)take((size_t)ROWS_ * C_ * 2);     // [B,H,HD,N]
    bf16*  attn   = (bf16*)take((size_t)ROWS_ * C_ * 2);     // [B,N,C]
    float* x1     = (float*)take((size_t)ROWS_ * C_ * 4);    // residual 1
    bf16*  h2     = (bf16*)take((size_t)ROWS_ * C_ * 2);     // ln2 out
    bf16*  fc1o   = (bf16*)take((size_t)ROWS_ * HID_ * 2);   // gelu(fc1)

    // 1) weight convert+transpose to bf16
    cvt_transpose_kernel<<<(3 * C_ * C_) / 256, 256, 0, stream>>>(w_qkv,  wqkvT,  C_,   3 * C_);
    cvt_transpose_kernel<<<(C_ * C_)     / 256, 256, 0, stream>>>(w_proj, wprojT, C_,   C_);
    cvt_transpose_kernel<<<(C_ * HID_)   / 256, 256, 0, stream>>>(w_fc1,  wfc1T,  C_,   HID_);
    cvt_transpose_kernel<<<(HID_ * C_)   / 256, 256, 0, stream>>>(w_fc2,  wfc2T,  HID_, C_);

    // 2) LN1
    ln_bf16_kernel<<<ROWS_, 256, 0, stream>>>(x, ln1_g, ln1_b, h1);

    // 3) QKV GEMM (M=8192, K=768, N=2304) -> scattered q/k/vT bf16
    gemm_bf16_kernel<0><<<dim3((3 * C_) / 128, ROWS_ / 128), 256, 0, stream>>>(
        h1, wqkvT, C_, 3 * C_, nullptr, nullptr, nullptr, nullptr, qbuf, kbuf, vbufT);

    // 4) flash attention -> attn bf16 [B,N,C]
    attn_kernel<<<(B_ * H_ * (N_ / 16)) / 4, 128, 0, stream>>>(qbuf, kbuf, vbufT, attn);

    // 5) proj GEMM + bias + residual -> x1 f32
    gemm_bf16_kernel<1><<<dim3(C_ / 128, ROWS_ / 128), 256, 0, stream>>>(
        attn, wprojT, C_, C_, b_proj, x, x1, nullptr, nullptr, nullptr, nullptr);

    // 6) LN2
    ln_bf16_kernel<<<ROWS_, 256, 0, stream>>>(x1, ln2_g, ln2_b, h2);

    // 7) FC1 GEMM + bias + exact GELU -> fc1o bf16
    gemm_bf16_kernel<2><<<dim3(HID_ / 128, ROWS_ / 128), 256, 0, stream>>>(
        h2, wfc1T, C_, HID_, b_fc1, nullptr, nullptr, fc1o, nullptr, nullptr, nullptr);

    // 8) FC2 GEMM + bias + residual -> d_out f32
    gemm_bf16_kernel<1><<<dim3(C_ / 128, ROWS_ / 128), 256, 0, stream>>>(
        fc1o, wfc2T, HID_, C_, b_fc2, x1, out, nullptr, nullptr, nullptr, nullptr);
}
